// TSAE_77841987272962
// MI455X (gfx1250) — matrix operand order repaired and checked by reading, not compiled, not run on hardware
//
#include <hip/hip_runtime.h>

typedef __attribute__((ext_vector_type(16))) _Float16 v16h;
typedef __attribute__((ext_vector_type(8)))  _Float16 h8;
typedef __attribute__((ext_vector_type(8)))  float    v8f;

#define N_NODES 8192
#define D_EMB   64

// ---------------- kernel 1: fp32 -> f16 convert + fp32 row squared norms ----
// One wave (32 lanes) per row of x (64 floats).
__global__ void __launch_bounds__(256)
prep_f16_norms(const float* __restrict__ x, _Float16* __restrict__ xh,
               float* __restrict__ sq) {
  const int wid  = (int)((blockIdx.x * blockDim.x + threadIdx.x) >> 5);
  const int lane = (int)(threadIdx.x & 31);
  if (wid >= N_NODES) return;
  const float* row = x + (size_t)wid * D_EMB;
  float a = row[lane];
  float b = row[lane + 32];
  _Float16* orow = xh + (size_t)wid * D_EMB;
  orow[lane]      = (_Float16)a;
  orow[lane + 32] = (_Float16)b;
  float s = __builtin_fmaf(a, a, b * b);
#pragma unroll
  for (int off = 16; off > 0; off >>= 1)
    s += __shfl_xor(s, off, 32);        // wave32 reduction
  if (lane == 0) sq[wid] = s;
}

// ---------------- kernel 2: WMMA x.x^T tile + distance/sigmoid epilogue -----
// Each wave computes a 16(M) x 32(N) strip of the adjacency matrix:
// 2 column tiles x 2 k-steps = 4x v_wmma_f32_16x16x32_f16.
__global__ void __launch_bounds__(256)
adj_wmma(const _Float16* __restrict__ xh, const float* __restrict__ sq,
         const float* __restrict__ Wp, const float* __restrict__ bp,
         float* __restrict__ out) {
  const int lane = (int)(threadIdx.x & 31);
  const int wid  = (int)((blockIdx.x * blockDim.x + threadIdx.x) >> 5);
  const int tilesN = N_NODES / 32;          // 256 strips per row band
  const int tm = wid / tilesN;              // 0..511  (M tile of 16)
  const int tn = wid - tm * tilesN;         // 0..255  (N strip of 32)
  const int m_base = tm * 16;
  const int n_base = tn * 32;
  const int half = lane >> 4;               // 0: lanes 0-15, 1: lanes 16-31
  const int l15  = lane & 15;

  // --- A fragments (16x32 f16, contiguous k = 16*half + s, same order as B):
  //   lanes 0-15 : lo8 = K[kk+0..7],   hi8 = K[kk+8..15]  of row m_base+l15
  //   lanes16-31 : lo8 = K[kk+16..23], hi8 = K[kk+24..31]
  const _Float16* arow = xh + (size_t)(m_base + l15) * D_EMB;
  h8 lo0 = *(const h8*)(arow + 0  + half * 16);
  h8 hi0 = *(const h8*)(arow + 8  + half * 16);
  h8 lo1 = *(const h8*)(arow + 32 + half * 16);
  h8 hi1 = *(const h8*)(arow + 40 + half * 16);
  v16h a0 = __builtin_shufflevector(lo0, hi0, 0,1,2,3,4,5,6,7,8,9,10,11,12,13,14,15);
  v16h a1 = __builtin_shufflevector(lo1, hi1, 0,1,2,3,4,5,6,7,8,9,10,11,12,13,14,15);

  // --- B fragments (32x16 f16): lane n holds 16 contiguous K values of
  // column row j = n_base(+16) + l15, offset kk + 16*half. B = x^T so this is
  // a contiguous 32-byte read of the f16 row.
  v8f acc0 = {};
  v8f acc1 = {};
  {
    const _Float16* brow = xh + (size_t)(n_base + l15) * D_EMB + half * 16;
    v16h b0 = *(const v16h*)(brow);
    v16h b1 = *(const v16h*)(brow + 32);
    acc0 = __builtin_amdgcn_wmma_f32_16x16x32_f16(false, a0, false, b0,
                                                  (short)0, acc0, false, false);
    acc0 = __builtin_amdgcn_wmma_f32_16x16x32_f16(false, a1, false, b1,
                                                  (short)0, acc0, false, false);
  }
  {
    const _Float16* brow = xh + (size_t)(n_base + 16 + l15) * D_EMB + half * 16;
    v16h b0 = *(const v16h*)(brow);
    v16h b1 = *(const v16h*)(brow + 32);
    acc1 = __builtin_amdgcn_wmma_f32_16x16x32_f16(false, a0, false, b0,
                                                  (short)0, acc1, false, false);
    acc1 = __builtin_amdgcn_wmma_f32_16x16x32_f16(false, a1, false, b1,
                                                  (short)0, acc1, false, false);
  }

  // --- epilogue: d2 = |xi|^2 + |xj|^2 - 2*dot ; sigmoid(W*d2+b); zero diag.
  // C layout: VGPR r -> M = r + 8*half, N = l15.
  const float Wv = Wp[0];
  const float bv = bp[0];
  const int   row_lo = m_base + half * 8;
  const int   col0 = n_base + l15;
  const int   col1 = n_base + 16 + l15;
  const float sqj0 = sq[col0];
  const float sqj1 = sq[col1];
#pragma unroll
  for (int r = 0; r < 8; ++r) {
    const int   row = row_lo + r;
    const float sqi = sq[row];
    {
      float d2  = sqi + sqj0 - 2.0f * acc0[r];
      float v   = __builtin_fmaf(Wv, d2, bv);
      float sgm = 1.0f / (1.0f + __expf(-v));
      if (row == col0) sgm = 0.0f;
      __builtin_nontemporal_store(sgm, out + (size_t)row * N_NODES + col0);
    }
    {
      float d2  = sqi + sqj1 - 2.0f * acc1[r];
      float v   = __builtin_fmaf(Wv, d2, bv);
      float sgm = 1.0f / (1.0f + __expf(-v));
      if (row == col1) sgm = 0.0f;
      __builtin_nontemporal_store(sgm, out + (size_t)row * N_NODES + col1);
    }
  }
}

extern "C" void kernel_launch(void* const* d_in, const int* in_sizes, int n_in,
                              void* d_out, int out_size, void* d_ws, size_t ws_size,
                              hipStream_t stream) {
  const float* x  = (const float*)d_in[0];
  const float* Wp = (const float*)d_in[1];
  const float* bp = (const float*)d_in[2];
  float* out = (float*)d_out;

  // workspace layout: [ xh : 8192*64 f16 = 1 MiB ][ sq : 8192 f32 = 32 KiB ]
  _Float16* xh = (_Float16*)d_ws;
  float*    sq = (float*)((char*)d_ws + (size_t)N_NODES * D_EMB * sizeof(_Float16));

  // kernel 1: 8192 waves, 8 waves per 256-thread block
  prep_f16_norms<<<N_NODES / 8, 256, 0, stream>>>(x, xh, sq);

  // kernel 2: (8192/16)*(8192/32) = 131072 waves -> 16384 blocks of 8 waves
  const int waves = (N_NODES / 16) * (N_NODES / 32);
  adj_wmma<<<waves / 8, 256, 0, stream>>>(xh, sq, Wp, bp, out);
}
